// ProbeMessageModel_42588895708032
// MI455X (gfx1250) — compile-verified, matrix operand order checked by reading, not run on hardware
//
#include <hip/hip_runtime.h>
#include <hip/hip_bf16.h>

typedef __attribute__((ext_vector_type(16))) _Float16 v16h;
typedef __attribute__((ext_vector_type(8)))  float    v8f;
typedef __attribute__((ext_vector_type(4)))  unsigned int u32x4;
typedef __attribute__((ext_vector_type(8)))  int      i32x8;
typedef __attribute__((ext_vector_type(4)))  int      i32x4;

#define NATOMS 20000
#define NPROBE 100000
#define NEDGE  400000
#define HD     128
#define NLAYER 3
#define WAVES  2

#if defined(__has_builtin)
#if __has_builtin(__builtin_amdgcn_tensor_load_to_lds)
#define HAVE_TDM 1
#endif
#endif
#ifndef HAVE_TDM
#define HAVE_TDM 0
#endif

__device__ __forceinline__ int imin(int a, int b) { return a < b ? a : b; }

__device__ __forceinline__ v8f zero8() {
  v8f z = {0.f, 0.f, 0.f, 0.f, 0.f, 0.f, 0.f, 0.f};
  return z;
}

__device__ __forceinline__ v8f wmma16(v16h a, v16h b, v8f c) {
  // D = A(16x32 f16) * B(32x16 f16) + C(16x16 f32)
  return __builtin_amdgcn_wmma_f32_16x16x32_f16(false, a, false, b, (short)0, c,
                                                false, false);
}

// Build a 16-bit A/B fragment for one lane from a row-major (row, K) matrix.
// A: row = m (lane&15); B (stored N-major, K-contiguous): row = n0 + (lane&15).
// ISA 16-bit 16x32 layout: lanes 0-15 hold K halves {0..7,16..23}+8*khi pattern.
__device__ __forceinline__ v16h load_frag(const _Float16* M, int ld, int row,
                                          int k0, int khi) {
  const _Float16* p = M + (size_t)row * ld + k0 + khi * 8;
  v16h f;
#pragma unroll
  for (int j = 0; j < 8; ++j) {
    const int kb = (j < 4) ? (2 * j) : (16 + 2 * (j - 4));
    f[2 * j]     = p[kb];
    f[2 * j + 1] = p[kb + 1];
  }
  return f;
}

// 16xK @ KxHD GEMM: A staged (LDS) f16, W pre-transposed N-major f16.
template <int KC>
__device__ __forceinline__ void gemm_16xH(const _Float16* A, int lda,
                                          const _Float16* W, int ldw,
                                          int col, int khi, v8f acc[8]) {
#pragma unroll
  for (int t = 0; t < 8; ++t) acc[t] = zero8();
#pragma unroll
  for (int kc = 0; kc < KC; ++kc) {
    v16h a = load_frag(A, lda, col, kc * 32, khi);
#pragma unroll
    for (int t = 0; t < 8; ++t) {
      v16h b = load_frag(W, ldw, t * 16 + col, kc * 32, khi);
      acc[t] = wmma16(a, b, acc[t]);
    }
  }
}

__device__ __forceinline__ float dssp(float x) {  // softplus(x) - log(2)
  return fmaxf(x, 0.f) + log1pf(__expf(-fabsf(x))) - 0.69314718055994531f;
}
__device__ __forceinline__ float dsig(float x) {
  return 1.f / (1.f + __expf(-x));
}

#if HAVE_TDM
// TDM: DMA a dense [nrows x rowlen] f32 tile (row stride = rowlen) from global
// memory into LDS. D# per CDNA5 ISA ch.8: group0 {count=1, lds_addr,
// global_addr, type=2}; group1 {data_size=4B, tensor dims/strides, tile dims}.
// Descriptor words forced wave-uniform via readfirstlane (SGPR groups).
__device__ __forceinline__ void tdm_load_rows_f32(void* lds_dst,
                                                  const float* gsrc,
                                                  int nrows, int rowlen) {
  unsigned int lds =
      __builtin_amdgcn_readfirstlane((unsigned int)(uintptr_t)lds_dst);
  unsigned long long ga = (unsigned long long)(uintptr_t)gsrc;
  unsigned int ga_lo = __builtin_amdgcn_readfirstlane((unsigned int)ga);
  unsigned int ga_hi = __builtin_amdgcn_readfirstlane((unsigned int)(ga >> 32));
  unsigned int rl = __builtin_amdgcn_readfirstlane((unsigned int)rowlen);
  unsigned int nr = __builtin_amdgcn_readfirstlane((unsigned int)nrows);

  u32x4 g0;
  g0[0] = 1u;                                    // count=1 (valid user D#)
  g0[1] = lds;                                   // lds_addr (bytes)
  g0[2] = ga_lo;                                 // global_addr[31:0]
  g0[3] = (ga_hi & 0x01FFFFFFu) | 0x80000000u;   // addr[56:32] | type=2

  i32x8 g1;
  g1[0] = (int)(2u << 16);                       // data_size=4B; no multicast
  g1[1] = (int)(rl << 16);                       // tensor_dim0[15:0] @bits63:48
  g1[2] = (int)((rl >> 16) | (nr << 16));        // dim0 hi | tensor_dim1 lo
  g1[3] = (int)(((nr >> 16) & 0xFFFFu) | (rl << 16));  // dim1 hi | tile_dim0
  g1[4] = (int)(nr & 0xFFFFu);                   // tile_dim1=nrows, tile_dim2=0
  g1[5] = (int)rl;                               // tensor_dim0_stride[31:0]
  g1[6] = 0;                                     // stride hi | dim1_stride lo
  g1[7] = 0;

  i32x4 gz = {0, 0, 0, 0};
#if __clang_major__ >= 23
  i32x8 gz8 = {0, 0, 0, 0, 0, 0, 0, 0};
  __builtin_amdgcn_tensor_load_to_lds(g0, g1, gz, gz, gz8, 0);
#else
  __builtin_amdgcn_tensor_load_to_lds(g0, g1, gz, gz, 0);
#endif
}
#endif  // HAVE_TDM

// --- weight prep: f32 [K][N] -> f16 [N][Kp] (K zero-padded to Kp) -------------
__global__ void transpose_f32_to_f16(const float* __restrict__ src,
                                     _Float16* __restrict__ dst,
                                     int K, int N, int Kp) {
  int idx = blockIdx.x * blockDim.x + threadIdx.x;
  if (idx >= N * Kp) return;
  int n = idx / Kp, k = idx - n * Kp;
  float v = (k < K) ? src[(size_t)k * N + n] : 0.f;
  dst[idx] = (_Float16)v;
}

// --- per-edge kernel: gate MLP, node MLP, scatter-add messages ---------------
__global__ __launch_bounds__(64) void edge_kernel(
    const float* __restrict__ atom_rep_l, const float* __restrict__ probe_state,
    const float* __restrict__ dist, const int* __restrict__ edges,
    const _Float16* __restrict__ We1t, const float* __restrict__ be1,
    const _Float16* __restrict__ We2t, const float* __restrict__ be2,
    const _Float16* __restrict__ Wn1t, const float* __restrict__ bn1,
    const _Float16* __restrict__ Wn2t, const float* __restrict__ bn2,
    float* __restrict__ msgsum) {
  __shared__ _Float16 s_nodes[WAVES][16 * 256];
  __shared__ _Float16 s_h[WAVES][16 * HD];
  __shared__ _Float16 s_g[WAVES][16 * HD];
  __shared__ float    s_sc[WAVES][16];
  __shared__ int      s_rcv[WAVES][16];

  const int wave = threadIdx.x >> 5;
  const int lane = threadIdx.x & 31;
  const int col = lane & 15;
  const int khi = lane >> 4;
  const int rowoff = khi * 8;
  const int e0 = (blockIdx.x * WAVES + wave) * 16;

  _Float16* nodes = s_nodes[wave];
  _Float16* hbuf = s_h[wave];
  _Float16* gbuf = s_g[wave];
  float* scb = s_sc[wave];
  int* rcvb = s_rcv[wave];

  // L2 prefetch of the streamed weight tables (global_prefetch_b8)
  if (lane == 0) {
    __builtin_prefetch(Wn1t, 0, 1);
    __builtin_prefetch(Wn2t, 0, 1);
  }

  // stage 0: radial basis edge features (K=16, zero-padded to 32), soft cutoff
  if (lane < 16) {
    int e = imin(e0 + lane, NEDGE - 1);
    float d = dist[e];
    scb[lane] = 1.0f - dsig(5.0f * (d - 2.5f));
    rcvb[lane] = edges[2 * e + 1];
#pragma unroll
    for (int k = 0; k < 16; ++k) {
      float t = d - 0.25f * (float)k;
      nodes[lane * 32 + k] = (_Float16)__expf(-8.0f * t * t);
      nodes[lane * 32 + 16 + k] = (_Float16)0.f;
    }
  }
  __syncthreads();

  v8f acc[8];

  // gate MLP layer 1: ssp(es @ We1 + be1)
  gemm_16xH<1>(nodes, 32, We1t, 32, col, khi, acc);
#pragma unroll
  for (int t = 0; t < 8; ++t) {
    const int n = t * 16 + col;
    const float bias = be1[n];
#pragma unroll
    for (int r = 0; r < 8; ++r)
      hbuf[(r + rowoff) * HD + n] = (_Float16)dssp(acc[t][r] + bias);
  }
  __syncthreads();

  // gate MLP layer 2, times soft_cut
  gemm_16xH<4>(hbuf, HD, We2t, HD, col, khi, acc);
#pragma unroll
  for (int t = 0; t < 8; ++t) {
    const int n = t * 16 + col;
    const float bias = be2[n];
#pragma unroll
    for (int r = 0; r < 8; ++r) {
      const int m = r + rowoff;
      gbuf[m * HD + n] = (_Float16)((acc[t][r] + bias) * scb[m]);
    }
  }
  __syncthreads();

  // gather concat(atom_rep[snd], probe_state[rcv]) -> f16 LDS (16 x 256)
  for (int m = 0; m < 16; ++m) {
    const int e = imin(e0 + m, NEDGE - 1);
    const int snd = edges[2 * e];
    const int rcv = edges[2 * e + 1];
    const int base = lane * 8;
    const float* src = (base < HD)
                           ? (atom_rep_l + (size_t)snd * HD + base)
                           : (probe_state + (size_t)rcv * HD + (base - HD));
    const float4 v0 = *(const float4*)(src);
    const float4 v1 = *(const float4*)(src + 4);
    _Float16* dst = nodes + m * 256 + base;
    dst[0] = (_Float16)v0.x; dst[1] = (_Float16)v0.y;
    dst[2] = (_Float16)v0.z; dst[3] = (_Float16)v0.w;
    dst[4] = (_Float16)v1.x; dst[5] = (_Float16)v1.y;
    dst[6] = (_Float16)v1.z; dst[7] = (_Float16)v1.w;
  }
  __syncthreads();

  // node MLP layer 1 (K=256)
  gemm_16xH<8>(nodes, 256, Wn1t, 256, col, khi, acc);
#pragma unroll
  for (int t = 0; t < 8; ++t) {
    const int n = t * 16 + col;
    const float bias = bn1[n];
#pragma unroll
    for (int r = 0; r < 8; ++r)
      hbuf[(r + rowoff) * HD + n] = (_Float16)dssp(acc[t][r] + bias);
  }
  __syncthreads();

  // node MLP layer 2 -> messages; gate; scatter-add segment sum (L2-resident)
  gemm_16xH<4>(hbuf, HD, Wn2t, HD, col, khi, acc);
#pragma unroll
  for (int t = 0; t < 8; ++t) {
    const int n = t * 16 + col;
    const float bias = bn2[n];
#pragma unroll
    for (int r = 0; r < 8; ++r) {
      const int m = r + rowoff;
      if (e0 + m < NEDGE) {
        const float msg = (acc[t][r] + bias) * (float)gbuf[m * HD + n];
        atomicAdd(msgsum + (size_t)rcvb[m] * HD + n, msg);
      }
    }
  }
}

// --- per-probe kernel: gated state update ------------------------------------
__global__ __launch_bounds__(64) void probe_kernel(
    float* __restrict__ probe_state, const float* __restrict__ msgsum,
    const _Float16* __restrict__ gW1t, const float* __restrict__ gb1,
    const _Float16* __restrict__ gW2t, const float* __restrict__ gb2,
    const _Float16* __restrict__ tW1t, const float* __restrict__ tb1,
    const _Float16* __restrict__ tW2t, const float* __restrict__ tb2) {
  __shared__ _Float16 s_in[WAVES][16 * HD];
  __shared__ _Float16 s_h[WAVES][16 * HD];
  __shared__ _Float16 s_gate[WAVES][16 * HD];
#if HAVE_TDM
  __shared__ float s_stage[WAVES][16 * HD];
#endif

  const int wave = threadIdx.x >> 5;
  const int lane = threadIdx.x & 31;
  const int col = lane & 15;
  const int khi = lane >> 4;
  const int rowoff = khi * 8;
  const int p0 = (blockIdx.x * WAVES + wave) * 16;

  _Float16* inb = s_in[wave];
  _Float16* hb = s_h[wave];
  _Float16* gb = s_gate[wave];

  // stage probe_state rows as f16
#if HAVE_TDM
  {
    float* fb = s_stage[wave];
    tdm_load_rows_f32(fb, probe_state + (size_t)p0 * HD, 16, HD);
    __builtin_amdgcn_s_wait_tensorcnt(0);
    for (int i = lane; i < 16 * HD; i += 32) inb[i] = (_Float16)fb[i];
  }
#else
  for (int m = 0; m < 16; ++m) {
    const int p = imin(p0 + m, NPROBE - 1);
    const float4 v = *(const float4*)(probe_state + (size_t)p * HD + lane * 4);
    _Float16* dst = inb + m * HD + lane * 4;
    dst[0] = (_Float16)v.x; dst[1] = (_Float16)v.y;
    dst[2] = (_Float16)v.z; dst[3] = (_Float16)v.w;
  }
#endif
  __syncthreads();

  v8f acc[8];

  // g = sigmoid(mlp(probe_state))
  gemm_16xH<4>(inb, HD, gW1t, HD, col, khi, acc);
#pragma unroll
  for (int t = 0; t < 8; ++t) {
    const int n = t * 16 + col;
    const float bias = gb1[n];
#pragma unroll
    for (int r = 0; r < 8; ++r)
      hb[(r + rowoff) * HD + n] = (_Float16)dssp(acc[t][r] + bias);
  }
  __syncthreads();
  gemm_16xH<4>(hb, HD, gW2t, HD, col, khi, acc);
#pragma unroll
  for (int t = 0; t < 8; ++t) {
    const int n = t * 16 + col;
    const float bias = gb2[n];
#pragma unroll
    for (int r = 0; r < 8; ++r)
      gb[(r + rowoff) * HD + n] = (_Float16)dsig(acc[t][r] + bias);
  }
  __syncthreads();

  // stage msgsum rows
#if HAVE_TDM
  {
    float* fb = s_stage[wave];
    tdm_load_rows_f32(fb, msgsum + (size_t)p0 * HD, 16, HD);
    __builtin_amdgcn_s_wait_tensorcnt(0);
    for (int i = lane; i < 16 * HD; i += 32) inb[i] = (_Float16)fb[i];
  }
#else
  for (int m = 0; m < 16; ++m) {
    const int p = imin(p0 + m, NPROBE - 1);
    const float4 v = *(const float4*)(msgsum + (size_t)p * HD + lane * 4);
    _Float16* dst = inb + m * HD + lane * 4;
    dst[0] = (_Float16)v.x; dst[1] = (_Float16)v.y;
    dst[2] = (_Float16)v.z; dst[3] = (_Float16)v.w;
  }
#endif
  __syncthreads();

  // t = mlp(msgsum); new_state = ps*g + (1-g)*t  (in-place, row-owned)
  gemm_16xH<4>(inb, HD, tW1t, HD, col, khi, acc);
#pragma unroll
  for (int t = 0; t < 8; ++t) {
    const int n = t * 16 + col;
    const float bias = tb1[n];
#pragma unroll
    for (int r = 0; r < 8; ++r)
      hb[(r + rowoff) * HD + n] = (_Float16)dssp(acc[t][r] + bias);
  }
  __syncthreads();
  gemm_16xH<4>(hb, HD, tW2t, HD, col, khi, acc);
#pragma unroll
  for (int t = 0; t < 8; ++t) {
    const int n = t * 16 + col;
    const float bias = tb2[n];
#pragma unroll
    for (int r = 0; r < 8; ++r) {
      const int m = r + rowoff;
      if (p0 + m < NPROBE) {
        const size_t idx = (size_t)(p0 + m) * HD + n;
        const float tv = acc[t][r] + bias;
        const float g = (float)gb[m * HD + n];
        const float ps = probe_state[idx];
        probe_state[idx] = ps * g + (1.f - g) * tv;
      }
    }
  }
}

// --- readout -----------------------------------------------------------------
__global__ __launch_bounds__(64) void readout_kernel(
    const float* __restrict__ probe_state, const _Float16* __restrict__ rW1t,
    const float* __restrict__ rb1, const float* __restrict__ rW2,
    const float* __restrict__ rb2, float* __restrict__ out) {
  __shared__ _Float16 s_in[WAVES][16 * HD];
  __shared__ float s_h[WAVES][16 * HD];
#if HAVE_TDM
  __shared__ float s_stage[WAVES][16 * HD];
#endif

  const int wave = threadIdx.x >> 5;
  const int lane = threadIdx.x & 31;
  const int col = lane & 15;
  const int khi = lane >> 4;
  const int rowoff = khi * 8;
  const int p0 = (blockIdx.x * WAVES + wave) * 16;

  _Float16* inb = s_in[wave];
  float* hb = s_h[wave];

#if HAVE_TDM
  {
    float* fb = s_stage[wave];
    tdm_load_rows_f32(fb, probe_state + (size_t)p0 * HD, 16, HD);
    __builtin_amdgcn_s_wait_tensorcnt(0);
    for (int i = lane; i < 16 * HD; i += 32) inb[i] = (_Float16)fb[i];
  }
#else
  for (int m = 0; m < 16; ++m) {
    const int p = imin(p0 + m, NPROBE - 1);
    const float4 v = *(const float4*)(probe_state + (size_t)p * HD + lane * 4);
    _Float16* dst = inb + m * HD + lane * 4;
    dst[0] = (_Float16)v.x; dst[1] = (_Float16)v.y;
    dst[2] = (_Float16)v.z; dst[3] = (_Float16)v.w;
  }
#endif
  __syncthreads();

  v8f acc[8];
  gemm_16xH<4>(inb, HD, rW1t, HD, col, khi, acc);
#pragma unroll
  for (int t = 0; t < 8; ++t) {
    const int n = t * 16 + col;
    const float bias = rb1[n];
#pragma unroll
    for (int r = 0; r < 8; ++r)
      hb[(r + rowoff) * HD + n] = dssp(acc[t][r] + bias);
  }
  __syncthreads();

  if (lane < 16) {
    const int p = p0 + lane;
    if (p < NPROBE) {
      float s = rb2[0];
      const float* hrow = hb + lane * HD;
#pragma unroll 8
      for (int n = 0; n < HD; ++n) s += hrow[n] * rW2[n];
      out[p] = s;
    }
  }
}

// --- host --------------------------------------------------------------------
static inline void tpose(const float* src, _Float16* dst, int K, int N, int Kp,
                         hipStream_t s) {
  int total = N * Kp;
  transpose_f32_to_f16<<<(total + 255) / 256, 256, 0, s>>>(src, dst, K, N, Kp);
}

extern "C" void kernel_launch(void* const* d_in, const int* in_sizes, int n_in,
                              void* d_out, int out_size, void* d_ws,
                              size_t ws_size, hipStream_t stream) {
  (void)in_sizes; (void)n_in; (void)out_size; (void)ws_size;

  const float* atom_rep = (const float*)d_in[0];
  const float* dist = (const float*)d_in[1];
  const float* We1 = (const float*)d_in[2];
  const float* be1 = (const float*)d_in[3];
  const float* We2 = (const float*)d_in[4];
  const float* be2 = (const float*)d_in[5];
  const float* Wn1 = (const float*)d_in[6];
  const float* bn1 = (const float*)d_in[7];
  const float* Wn2 = (const float*)d_in[8];
  const float* bn2 = (const float*)d_in[9];
  const float* gW1 = (const float*)d_in[10];
  const float* gb1 = (const float*)d_in[11];
  const float* gW2 = (const float*)d_in[12];
  const float* gb2 = (const float*)d_in[13];
  const float* tW1 = (const float*)d_in[14];
  const float* tb1 = (const float*)d_in[15];
  const float* tW2 = (const float*)d_in[16];
  const float* tb2 = (const float*)d_in[17];
  const float* rW1 = (const float*)d_in[18];
  const float* rb1 = (const float*)d_in[19];
  const float* rW2 = (const float*)d_in[20];
  const float* rb2 = (const float*)d_in[21];
  const int* edges = (const int*)d_in[22];
  float* out = (float*)d_out;

  float* probe_state = (float*)d_ws;
  float* msgsum = probe_state + (size_t)NPROBE * HD;
  _Float16* wb = (_Float16*)(msgsum + (size_t)NPROBE * HD);

  const size_t S_We1 = 128 * 32, S_We2 = 128 * 128, S_Wn1 = 128 * 256,
               S_Sq = 128 * 128;
  const size_t O_We2 = S_We1;
  const size_t O_Wn1 = O_We2 + S_We2;
  const size_t O_Wn2 = O_Wn1 + S_Wn1;
  const size_t O_g1 = O_Wn2 + S_Sq;
  const size_t O_g2 = O_g1 + S_Sq;
  const size_t O_t1 = O_g2 + S_Sq;
  const size_t O_t2 = O_t1 + S_Sq;
  const size_t PER_LAYER = O_t2 + S_Sq;

  for (int l = 0; l < NLAYER; ++l) {
    _Float16* base = wb + (size_t)l * PER_LAYER;
    tpose(We1 + (size_t)l * 16 * HD, base, 16, HD, 32, stream);
    tpose(We2 + (size_t)l * HD * HD, base + O_We2, HD, HD, HD, stream);
    tpose(Wn1 + (size_t)l * 2 * HD * HD, base + O_Wn1, 2 * HD, HD, 2 * HD, stream);
    tpose(Wn2 + (size_t)l * HD * HD, base + O_Wn2, HD, HD, HD, stream);
    tpose(gW1 + (size_t)l * HD * HD, base + O_g1, HD, HD, HD, stream);
    tpose(gW2 + (size_t)l * HD * HD, base + O_g2, HD, HD, HD, stream);
    tpose(tW1 + (size_t)l * HD * HD, base + O_t1, HD, HD, HD, stream);
    tpose(tW2 + (size_t)l * HD * HD, base + O_t2, HD, HD, HD, stream);
  }
  _Float16* rW1t = wb + (size_t)NLAYER * PER_LAYER;
  tpose(rW1, rW1t, HD, HD, HD, stream);

  hipMemsetAsync(probe_state, 0, (size_t)NPROBE * HD * sizeof(float), stream);

  for (int l = 0; l < NLAYER; ++l) {
    hipMemsetAsync(msgsum, 0, (size_t)NPROBE * HD * sizeof(float), stream);
    _Float16* base = wb + (size_t)l * PER_LAYER;
    edge_kernel<<<NEDGE / (16 * WAVES), 64, 0, stream>>>(
        atom_rep + (size_t)l * NATOMS * HD, probe_state, dist, edges,
        base, be1 + l * HD, base + O_We2, be2 + l * HD,
        base + O_Wn1, bn1 + l * HD, base + O_Wn2, bn2 + l * HD, msgsum);
    probe_kernel<<<NPROBE / (16 * WAVES), 64, 0, stream>>>(
        probe_state, msgsum, base + O_g1, gb1 + l * HD, base + O_g2,
        gb2 + l * HD, base + O_t1, tb1 + l * HD, base + O_t2, tb2 + l * HD);
  }
  readout_kernel<<<NPROBE / (16 * WAVES), 64, 0, stream>>>(
      probe_state, rW1t, rb1, rW2, rb2, out);
}
